// GCN_16346645529165
// MI455X (gfx1250) — matrix operand order repaired and checked by reading, not compiled, not run on hardware
//
#include <hip/hip_runtime.h>

#define N_FEAT    512
#define HIDDEN    128
#define N_CLASSES 64

typedef __attribute__((ext_vector_type(16))) __bf16         v16bf;
typedef __attribute__((ext_vector_type(8)))  float          v8f;
typedef __attribute__((ext_vector_type(8)))  unsigned int   v8u;

struct U8x32 { uint4 lo; uint4 hi; };   // 32 bytes = 16 bf16

__device__ __forceinline__ unsigned short f32_to_bf16(float f) {
    union { float f; unsigned int u; } cv; cv.f = f;
    unsigned int u = cv.u;
    u += 0x7FFFu + ((u >> 16) & 1u);          // round-to-nearest-even
    return (unsigned short)(u >> 16);
}
__device__ __forceinline__ unsigned int pack2_bf16(float a, float b) {
    return (unsigned int)f32_to_bf16(a) | ((unsigned int)f32_to_bf16(b) << 16);
}

// ---------------------------------------------------------------------------
// Pack W[Ktot,Ntot] (f32, row-major) into fragment-major bf16 tiles.
// Tile (kt,nt): lane L holds 16 bf16 { K = kt*32 + 16*(L>=16) + i, N = nt*16 + (L&15) }
// stored contiguously at Wp[((kt*ntiles+nt)*32 + L)*16].  One wave per tile.
// ---------------------------------------------------------------------------
__global__ __launch_bounds__(256) void gcn_packW(const float* __restrict__ W,
                                                 unsigned short* __restrict__ Wp,
                                                 int Ktot, int Ntot) {
    const int lane  = threadIdx.x & 31;
    const int wave  = threadIdx.x >> 5;
    const int ntiles = Ntot >> 4;
    const int tiles  = (Ktot >> 5) * ntiles;
    const int tile   = blockIdx.x * 8 + wave;
    if (tile >= tiles) return;
    const int kt = tile / ntiles, nt = tile % ntiles;
    const int koff = kt * 32 + ((lane >> 4) << 4);
    const int n    = nt * 16 + (lane & 15);

    unsigned int outw[8];
#pragma unroll
    for (int i = 0; i < 8; ++i)
        outw[i] = pack2_bf16(W[(size_t)(koff + 2 * i) * Ntot + n],
                             W[(size_t)(koff + 2 * i + 1) * Ntot + n]);
    uint4* dst = (uint4*)(Wp + ((size_t)tile * 32 + lane) * 16);
    dst[0] = make_uint4(outw[0], outw[1], outw[2], outw[3]);
    dst[1] = make_uint4(outw[4], outw[5], outw[6], outw[7]);
}

// ---------------------------------------------------------------------------
// GEMM1: h[M,128] = x[M,512] @ W1  (A staged f32->bf16 through LDS, B pre-packed)
// block = 256 thr = 8 waves; block -> 16 rows, wave -> one 16-col N tile.
// ---------------------------------------------------------------------------
__global__ __launch_bounds__(256) void gcn_gemm1(const float* __restrict__ x,
                                                 const unsigned short* __restrict__ W1p,
                                                 float* __restrict__ h) {
    __shared__ __align__(16) unsigned short As[16 * N_FEAT];   // 16 KB bf16 panel

    const int t     = threadIdx.x;
    const int lane  = t & 31;
    const int wave  = t >> 5;                 // N tile 0..7
    const int half  = lane >> 4;
    const int l16   = lane & 15;
    const int mBase = blockIdx.x * 16;

    // ---- stage & convert A panel: 8192 floats, 32 per thread -------------
    {
        const float4* xs = (const float4*)(x + (size_t)mBase * N_FEAT) + t * 8;
        unsigned int outw[16];
#pragma unroll
        for (int j = 0; j < 8; ++j) {
            float4 v = xs[j];
            outw[2 * j]     = pack2_bf16(v.x, v.y);
            outw[2 * j + 1] = pack2_bf16(v.z, v.w);
        }
        uint4* dst = (uint4*)As + t * 4;
#pragma unroll
        for (int j = 0; j < 4; ++j)
            dst[j] = make_uint4(outw[4 * j], outw[4 * j + 1],
                                outw[4 * j + 2], outw[4 * j + 3]);
    }
    __syncthreads();

    v8f acc = {};
    const unsigned short* arow = As + (size_t)l16 * N_FEAT;
#pragma unroll 4
    for (int kk = 0; kk < N_FEAT; kk += 32) {
        const int ka = kk + half * 16;
        U8x32 a8;
        a8.lo = *(const uint4*)(arow + ka);        // K ka..ka+7
        a8.hi = *(const uint4*)(arow + ka + 8);    // K ka+8..ka+15
        v8u b8 = *(const v8u*)(W1p + (((size_t)(kk >> 5) * 8 + wave) * 32 + lane) * 16);

        v16bf a = __builtin_bit_cast(v16bf, a8);
        v16bf b = __builtin_bit_cast(v16bf, b8);
        acc = __builtin_amdgcn_wmma_f32_16x16x32_bf16(false, a, false, b,
                                                      (short)0, acc, false, false);
    }

    float* crow = h + (size_t)mBase * HIDDEN + wave * 16 + l16;
#pragma unroll
    for (int r = 0; r < 8; ++r)
        crow[(size_t)(r + 8 * half) * HIDDEN] = acc[r];
}

// ---------------------------------------------------------------------------
// GEMM2: h3[M,64] = relu(p1[M,128] + b1) @ W2   (relu+bias fused into staging)
// block = 8 waves = 2 M-tiles x 4 N-tiles -> 32 rows x 64 cols per block.
// ---------------------------------------------------------------------------
__global__ __launch_bounds__(256) void gcn_gemm2(const float* __restrict__ p1,
                                                 const float* __restrict__ b1,
                                                 const unsigned short* __restrict__ W2p,
                                                 float* __restrict__ h3) {
    __shared__ __align__(16) unsigned short As[32 * HIDDEN];   // 8 KB bf16 panel

    const int t     = threadIdx.x;
    const int lane  = t & 31;
    const int wave  = t >> 5;
    const int nt    = wave & 3;
    const int mt    = wave >> 2;
    const int half  = lane >> 4;
    const int l16   = lane & 15;
    const int mBase = blockIdx.x * 32;

    // ---- stage relu(p1+b1) panel: 4096 floats, 16 per thread -------------
    {
        const int row = t >> 3;
        const int k0  = (t & 7) * 16;
        const float4* pr = (const float4*)(p1 + (size_t)(mBase + row) * HIDDEN + k0);
        const float4* br = (const float4*)(b1 + k0);
        unsigned int outw[8];
#pragma unroll
        for (int j = 0; j < 4; ++j) {
            float4 v = pr[j], bb = br[j];
            float e0 = v.x + bb.x, e1 = v.y + bb.y, e2 = v.z + bb.z, e3 = v.w + bb.w;
            e0 = e0 > 0.f ? e0 : 0.f;  e1 = e1 > 0.f ? e1 : 0.f;
            e2 = e2 > 0.f ? e2 : 0.f;  e3 = e3 > 0.f ? e3 : 0.f;
            outw[2 * j]     = pack2_bf16(e0, e1);
            outw[2 * j + 1] = pack2_bf16(e2, e3);
        }
        uint4* dst = (uint4*)(As + (size_t)row * HIDDEN + k0);
        dst[0] = make_uint4(outw[0], outw[1], outw[2], outw[3]);
        dst[1] = make_uint4(outw[4], outw[5], outw[6], outw[7]);
    }
    __syncthreads();

    v8f acc = {};
    const unsigned short* arow = As + (size_t)(mt * 16 + l16) * HIDDEN;
#pragma unroll
    for (int kk = 0; kk < HIDDEN; kk += 32) {
        const int ka = kk + half * 16;
        U8x32 a8;
        a8.lo = *(const uint4*)(arow + ka);
        a8.hi = *(const uint4*)(arow + ka + 8);
        v8u b8 = *(const v8u*)(W2p + (((size_t)(kk >> 5) * 4 + nt) * 32 + lane) * 16);

        v16bf a = __builtin_bit_cast(v16bf, a8);
        v16bf b = __builtin_bit_cast(v16bf, b8);
        acc = __builtin_amdgcn_wmma_f32_16x16x32_bf16(false, a, false, b,
                                                      (short)0, acc, false, false);
    }

    float* crow = h3 + (size_t)(mBase + mt * 16) * N_CLASSES + nt * 16 + l16;
#pragma unroll
    for (int r = 0; r < 8; ++r)
        crow[(size_t)(r + 8 * half) * N_CLASSES] = acc[r];
}

// ---------------------------------------------------------------------------
// Edge scatter: out[dst] += w * h[src]   (one wave per edge; L2-resident atomics)
// ---------------------------------------------------------------------------
__global__ __launch_bounds__(256) void gcn_scatter128(const float* __restrict__ h,
                                                      const int* __restrict__ src,
                                                      const int* __restrict__ dst,
                                                      const float* __restrict__ ew,
                                                      float* __restrict__ out, int E) {
    const int e = blockIdx.x * 8 + (threadIdx.x >> 5);
    if (e >= E) return;
    const int lane = threadIdx.x & 31;
    const int s = src[e];
    const int d = dst[e];
    const float w = ew[e];
    float4 v = ((const float4*)(h + (size_t)s * HIDDEN))[lane];
    float* o = out + (size_t)d * HIDDEN + lane * 4;
    atomicAdd(o + 0, v.x * w);
    atomicAdd(o + 1, v.y * w);
    atomicAdd(o + 2, v.z * w);
    atomicAdd(o + 3, v.w * w);
}

__global__ __launch_bounds__(256) void gcn_scatter64(const float* __restrict__ h3,
                                                     const int* __restrict__ src,
                                                     const int* __restrict__ dst,
                                                     const float* __restrict__ ew,
                                                     float* __restrict__ out, int E) {
    const int e = blockIdx.x * 8 + (threadIdx.x >> 5);
    if (e >= E) return;
    const int lane = threadIdx.x & 31;
    const int s = src[e];
    const int d = dst[e];
    const float w = ew[e];
    float2 v = ((const float2*)(h3 + (size_t)s * N_CLASSES))[lane];
    float* o = out + (size_t)d * N_CLASSES + lane * 2;
    atomicAdd(o + 0, v.x * w);
    atomicAdd(o + 1, v.y * w);
}

// ---------------------------------------------------------------------------
__global__ void gcn_zero(float* __restrict__ p, long long n) {
    long long i = (long long)blockIdx.x * blockDim.x + threadIdx.x;
    if (i < n) p[i] = 0.0f;
}
__global__ void gcn_init_b2(float* __restrict__ out, const float* __restrict__ b2,
                            long long n) {
    long long i = (long long)blockIdx.x * blockDim.x + threadIdx.x;
    if (i < n) out[i] = b2[i & (N_CLASSES - 1)];
}

// ---------------------------------------------------------------------------
extern "C" void kernel_launch(void* const* d_in, const int* in_sizes, int n_in,
                              void* d_out, int out_size, void* d_ws, size_t ws_size,
                              hipStream_t stream) {
    const float* x  = (const float*)d_in[0];
    const int*   ei = (const int*)d_in[1];
    const float* ew = (const float*)d_in[2];
    const float* W1 = (const float*)d_in[3];
    const float* b1 = (const float*)d_in[4];
    const float* W2 = (const float*)d_in[5];
    const float* b2 = (const float*)d_in[6];
    float* out = (float*)d_out;

    const int E = in_sizes[2];              // 1,600,000
    const int M = in_sizes[0] / N_FEAT;     // 100,000

    const int* src = ei;
    const int* dst = ei + E;

    char* ws = (char*)d_ws;
    const size_t szH  = (size_t)M * HIDDEN * sizeof(float);      // 51.2 MB
    const size_t szH3 = (size_t)M * N_CLASSES * sizeof(float);   // 25.6 MB
    float*          h   = (float*)ws;
    float*          p1  = (float*)(ws + szH);
    float*          h3  = (float*)(ws + 2 * szH);
    unsigned short* W1p = (unsigned short*)(ws + 2 * szH + szH3);
    unsigned short* W2p = (unsigned short*)(ws + 2 * szH + szH3
                                            + (size_t)N_FEAT * HIDDEN * 2);

    const long long nP1  = (long long)M * HIDDEN;
    const long long nOut = (long long)M * N_CLASSES;

    // pre-pack weights to fragment-major bf16 (L2 resident)
    gcn_packW<<<(N_FEAT / 32) * (HIDDEN / 16) / 8, 256, 0, stream>>>(W1, W1p, N_FEAT, HIDDEN);
    gcn_packW<<<(HIDDEN / 32) * (N_CLASSES / 16) / 8, 256, 0, stream>>>(W2, W2p, HIDDEN, N_CLASSES);

    // h = x @ W1
    gcn_gemm1<<<M / 16, 256, 0, stream>>>(x, W1p, h);
    // p1 = 0 ; p1[dst] += w * h[src]
    gcn_zero<<<(unsigned)((nP1 + 255) / 256), 256, 0, stream>>>(p1, nP1);
    gcn_scatter128<<<(E + 7) / 8, 256, 0, stream>>>(h, src, dst, ew, p1, E);
    // h3 = relu(p1 + b1) @ W2
    gcn_gemm2<<<M / 32, 256, 0, stream>>>(p1, b1, W2p, h3);
    // out = b2 ; out[dst] += w * h3[src]
    gcn_init_b2<<<(unsigned)((nOut + 255) / 256), 256, 0, stream>>>(out, b2, nOut);
    gcn_scatter64<<<(E + 7) / 8, 256, 0, stream>>>(h3, src, dst, ew, out, E);
}
